// MHAttention_85152021610643
// MI455X (gfx1250) — compile-verified, matrix operand order checked by reading, not run on hardware
//
#include <hip/hip_runtime.h>
#include <hip/hip_bf16.h>
#include <math.h>

typedef _Float16 half_t;
typedef __attribute__((ext_vector_type(16))) _Float16 v16h;
typedef __attribute__((ext_vector_type(8)))  _Float16 v8h;
typedef __attribute__((ext_vector_type(8)))  float    v8f;
typedef __attribute__((ext_vector_type(4)))  float    v4f;
typedef __attribute__((ext_vector_type(4)))  unsigned int v4u;
typedef __attribute__((ext_vector_type(8)))  int      v8i;
typedef __attribute__((ext_vector_type(4)))  int      v4i;

#define SEQ    1024
#define DMODEL 1024
#define NHEADS 16
#define DEPTH  64
#define BATCH  4
#define MTOT  (BATCH * SEQ)   // 4096 flattened rows

// ---------------------------------------------------------------------------
// helpers: load 8 consecutive f32 and pack into halves [base..base+7] of v16h
// ---------------------------------------------------------------------------
__device__ __forceinline__ void load8_f32_cvt(const float* __restrict__ p,
                                              v16h& dst, int base) {
  v4f x0 = *reinterpret_cast<const v4f*>(p);
  v4f x1 = *reinterpret_cast<const v4f*>(p + 4);
#pragma unroll
  for (int i = 0; i < 4; ++i) {
    dst[base + i]     = (_Float16)x0[i];
    dst[base + 4 + i] = (_Float16)x1[i];
  }
}

// load 8 consecutive f16 (one 16-byte vector) into halves [base..base+7]
__device__ __forceinline__ void load8_f16(const half_t* __restrict__ p,
                                          v16h& dst, int base) {
  v8h x = *reinterpret_cast<const v8h*>(p);
#pragma unroll
  for (int i = 0; i < 8; ++i) dst[base + i] = x[i];
}

// ---------------------------------------------------------------------------
// TDM: async DMA of a 2D f16 tile (32 rows x 64 halves, row stride 1024
// halves) from global memory into LDS.  D# packing per CDNA5 ISA ch.8:
//   group0: [1:0]=count=1, [63:32]=lds_addr, [120:64]=global_addr,
//           [127:126]=type=2
//   group1: [17:16]=data_size(1 => 2B), [79:48]=tensor_dim0=64,
//           [111:80]=tensor_dim1=32, [127:112]=tile_dim0=64,
//           [143:128]=tile_dim1=32, [207:160]=tensor_dim0_stride=1024
//   groups2/3: zero (2D tensor)
// Tracked by TENSORcnt; EXEC ignored (one DMA per instruction).
// ---------------------------------------------------------------------------
__device__ __forceinline__ void tdm_load_v_chunk(const half_t* gsrc,
                                                 unsigned lds_byte_off) {
  unsigned long long ga = (unsigned long long)(uintptr_t)gsrc;
  v4u g0;
  g0[0] = 1u;                                   // count=1 (valid user D#)
  g0[1] = lds_byte_off;                         // lds_addr (bytes)
  g0[2] = (unsigned)(ga & 0xFFFFFFFFu);         // global_addr[31:0]
  g0[3] = (unsigned)((ga >> 32) & 0x01FFFFFFu)  // global_addr[56:32]
          | (2u << 30);                         // type = 2 ("image")
  v8i g1;
  g1[0] = (int)(1u << 16);    // workgroup_mask=0, data_size=1 (2 bytes)
  g1[1] = (int)(64u << 16);   // tensor_dim0[15:0]=64 in dw1[31:16]
  g1[2] = (int)(32u << 16);   // tensor_dim0 hi=0, tensor_dim1[15:0]=32
  g1[3] = (int)(64u << 16);   // tensor_dim1 hi=0, tile_dim0=64
  g1[4] = (int)(32u);         // tile_dim1=32, tile_dim2=0
  g1[5] = (int)(1024u);       // tensor_dim0_stride[31:0] = 1024 elements
  g1[6] = 0;                  // stride hi / tensor_dim1_stride lo
  g1[7] = 0;                  // tensor_dim1_stride hi
  v4i gz = {0, 0, 0, 0};
#if defined(__clang_major__) && (__clang_major__ >= 23)
  v8i gz8 = {0, 0, 0, 0, 0, 0, 0, 0};
  __builtin_amdgcn_tensor_load_to_lds(g0, g1, gz, gz, gz8, 0);
#else
  __builtin_amdgcn_tensor_load_to_lds(g0, g1, gz, gz, 0);
#endif
}

// ---------------------------------------------------------------------------
// Kernel 1: Y[M,N](f16) = X[M,K](f32) @ W[N,K]^T (f32) + bias  (Q/K/V proj)
// One wave32 per 16(M) x 64(N) strip; A-fragment reused across 4 N-tiles.
// Fragment layout per CDNA5 ISA: lane L -> row = L&15, halves h map to
// K = h + (h>=8 ? 8 : 0) + ((L>=16) ? 8 : 0)  (two contiguous 8-elt runs).
// ---------------------------------------------------------------------------
__global__ __launch_bounds__(32)
void proj_gemm_f32(const float* __restrict__ X, const float* __restrict__ W,
                   const float* __restrict__ bias, half_t* __restrict__ Y,
                   int M, int N, int K) {
  const int n0   = blockIdx.x * 64;
  const int m0   = blockIdx.y * 16;
  const int lane = threadIdx.x;
  const int row  = lane & 15;
  const int hi   = lane >> 4;
  const int off  = hi * 8;

  v8f acc[4] = {};
  const float* xrow = X + (size_t)(m0 + row) * K;

  for (int k0 = 0; k0 < K; k0 += 32) {
    v16h a;
    load8_f32_cvt(xrow + k0 + off,      a, 0);
    load8_f32_cvt(xrow + k0 + 16 + off, a, 8);
#pragma unroll
    for (int t = 0; t < 4; ++t) {
      const float* wrow = W + (size_t)(n0 + t * 16 + row) * K;
      v16h b;
      load8_f32_cvt(wrow + k0 + off,      b, 0);
      load8_f32_cvt(wrow + k0 + 16 + off, b, 8);
      acc[t] = __builtin_amdgcn_wmma_f32_16x16x32_f16(
          false, a, false, b, (short)0, acc[t], false, false);
    }
  }
#pragma unroll
  for (int t = 0; t < 4; ++t) {
    const float bn = bias[n0 + t * 16 + row];  // C layout: lane's N = lane&15
#pragma unroll
    for (int r = 0; r < 8; ++r) {
      const int m = m0 + r + hi * 8;
      Y[(size_t)m * N + n0 + t * 16 + row] = (half_t)(acc[t][r] + bn);
    }
  }
}

// ---------------------------------------------------------------------------
// Kernel 2: flash attention. One wave32 per (b, h, 16 q-rows).
// V chunks streamed into double-buffered LDS by the Tensor Data Mover
// (tensor_load_to_lds + s_wait_tensorcnt) while the logits WMMAs run.
// Per 32-key chunk: 2x2 WMMA for logits, online softmax (shfl_xor butterfly
// within each 16-lane half + v_exp), P -> LDS -> A-fragment, 4 WMMAs for P@V.
// mask (1,1,S,S) applied; it lives in L2 (4MB << 192MB).
// ---------------------------------------------------------------------------
__global__ __launch_bounds__(32)
void attn_kernel(const half_t* __restrict__ Qp, const half_t* __restrict__ Kp,
                 const half_t* __restrict__ Vp, const float* __restrict__ mask,
                 half_t* __restrict__ Op) {
  __shared__ half_t lds_v[2][32 * DEPTH];  // double-buffered V chunk [key][depth]
  __shared__ half_t lds_p[16 * 32];        // probs tile [m][key-in-chunk]

  const int lane = threadIdx.x;
  const int tile = blockIdx.x;             // b*H*(S/16) tiles
  const int qt   = tile & (SEQ / 16 - 1);
  const int h    = (tile >> 6) & (NHEADS - 1);
  const int b    = tile >> 10;
  const int m0   = qt * 16;
  const int row  = lane & 15;
  const int hi   = lane >> 4;
  const int off  = hi * 8;
  const float scale = 0.125f;              // 1/sqrt(DEPTH)

  const unsigned lds_v_off[2] = {
      (unsigned)(uintptr_t)(&lds_v[0][0]),
      (unsigned)(uintptr_t)(&lds_v[1][0])};
  const half_t* vhead = Vp + (size_t)b * SEQ * DMODEL + h * DEPTH;

  // Q A-fragments for the two 32-wide depth chunks
  v16h qa[2];
  const half_t* qrow = Qp + ((size_t)(b * SEQ + m0 + row)) * DMODEL + h * DEPTH;
#pragma unroll
  for (int c = 0; c < 2; ++c) {
    load8_f16(qrow + c * 32 + off,      qa[c], 0);
    load8_f16(qrow + c * 32 + 16 + off, qa[c], 8);
  }

  // prologue: TDM-stream first V chunk into buffer 0
  tdm_load_v_chunk(vhead, lds_v_off[0]);

  float m_i[8], l_i[8];
#pragma unroll
  for (int r = 0; r < 8; ++r) { m_i[r] = -1e30f; l_i[r] = 0.0f; }
  v8f o[4] = {};

  for (int j0 = 0; j0 < SEQ; j0 += 32) {
    const int buf = (j0 >> 5) & 1;

    // kick the TDM for the NEXT V chunk into the other buffer; prefetch K
    if (j0 + 32 < SEQ) {
      tdm_load_v_chunk(vhead + (size_t)(j0 + 32) * DMODEL, lds_v_off[buf ^ 1]);
      __builtin_prefetch(
          Kp + ((size_t)(b * SEQ + j0 + 32 + lane)) * DMODEL + h * DEPTH, 0, 1);
    }

    // logits tiles: s[t] = Q(16x64) @ K(16 keys x 64)^T  (overlaps the DMA)
    v8f s[2];
#pragma unroll
    for (int t = 0; t < 2; ++t) {
      const half_t* krow =
          Kp + ((size_t)(b * SEQ + j0 + t * 16 + row)) * DMODEL + h * DEPTH;
      v8f z = {};
#pragma unroll
      for (int c = 0; c < 2; ++c) {
        v16h kb;
        load8_f16(krow + c * 32 + off,      kb, 0);
        load8_f16(krow + c * 32 + 16 + off, kb, 8);
        z = __builtin_amdgcn_wmma_f32_16x16x32_f16(
            false, qa[c], false, kb, (short)0, z, false, false);
      }
      s[t] = z;
    }

    // online softmax over this 32-key chunk
    float p0v[8], p1v[8], corr[8];
#pragma unroll
    for (int r = 0; r < 8; ++r) {
      const int m = m0 + r + hi * 8;
      float s0 = s[0][r] * scale + mask[(size_t)m * SEQ + j0 + row]      * (-1e9f);
      float s1 = s[1][r] * scale + mask[(size_t)m * SEQ + j0 + 16 + row] * (-1e9f);
      float mx = fmaxf(s0, s1);
#pragma unroll
      for (int d = 8; d >= 1; d >>= 1) mx = fmaxf(mx, __shfl_xor(mx, d, 32));
      const float mnew = fmaxf(m_i[r], mx);
      const float cexp = __expf(m_i[r] - mnew);
      const float p0   = __expf(s0 - mnew);
      const float p1   = __expf(s1 - mnew);
      float rs = p0 + p1;
#pragma unroll
      for (int d = 8; d >= 1; d >>= 1) rs += __shfl_xor(rs, d, 32);
      l_i[r] = l_i[r] * cexp + rs;
      m_i[r] = mnew;
      corr[r] = cexp;
      p0v[r] = p0; p1v[r] = p1;
    }

    // P (f16) -> LDS for cross-lane C-layout -> A-layout transpose
#pragma unroll
    for (int r = 0; r < 8; ++r) {
      lds_p[(r + hi * 8) * 32 + row]      = (half_t)p0v[r];
      lds_p[(r + hi * 8) * 32 + 16 + row] = (half_t)p1v[r];
    }
    __syncthreads();

    v16h pa;
    {
      const half_t* pr = &lds_p[row * 32];
      load8_f16(pr + off,      pa, 0);
      load8_f16(pr + 16 + off, pa, 8);
    }

    // current V chunk must be resident: <=1 outstanding leaves only the
    // next-buffer DMA in flight (TENSORcnt decrements in order)
    if (j0 + 32 < SEQ) __builtin_amdgcn_s_wait_tensorcnt(1);
    else               __builtin_amdgcn_s_wait_tensorcnt(0);

    // o[dt] = corr*o[dt] + P(16x32) @ V(32 x 16-depth-tile)
    const half_t* vbuf = &lds_v[buf][0];
#pragma unroll
    for (int dt = 0; dt < 4; ++dt) {
      v16h vb;
#pragma unroll
      for (int hh = 0; hh < 16; ++hh) {
        const int k = hh + (hh >= 8 ? 8 : 0) + off;  // key within chunk
        vb[hh] = vbuf[k * DEPTH + dt * 16 + row];
      }
      v8f oo = o[dt];
#pragma unroll
      for (int r = 0; r < 8; ++r) oo[r] *= corr[r];
      o[dt] = __builtin_amdgcn_wmma_f32_16x16x32_f16(
          false, pa, false, vb, (short)0, oo, false, false);
    }
  }

  // normalize + store attn (already in [B, S, H*depth] = concat-head layout)
#pragma unroll
  for (int dt = 0; dt < 4; ++dt)
#pragma unroll
    for (int r = 0; r < 8; ++r) {
      const int m = m0 + r + hi * 8;
      Op[((size_t)(b * SEQ + m)) * DMODEL + h * DEPTH + dt * 16 + row] =
          (half_t)(o[dt][r] / l_i[r]);
    }
}

// ---------------------------------------------------------------------------
// Kernel 3: out[M,N](f32) = A[M,K](f16) @ W[N,K]^T (f32) + bias  (final proj)
// ---------------------------------------------------------------------------
__global__ __launch_bounds__(32)
void out_gemm_f16(const half_t* __restrict__ X, const float* __restrict__ W,
                  const float* __restrict__ bias, float* __restrict__ Y,
                  int M, int N, int K) {
  const int n0   = blockIdx.x * 64;
  const int m0   = blockIdx.y * 16;
  const int lane = threadIdx.x;
  const int row  = lane & 15;
  const int hi   = lane >> 4;
  const int off  = hi * 8;

  v8f acc[4] = {};
  const half_t* xrow = X + (size_t)(m0 + row) * K;

  for (int k0 = 0; k0 < K; k0 += 32) {
    v16h a;
    load8_f16(xrow + k0 + off,      a, 0);
    load8_f16(xrow + k0 + 16 + off, a, 8);
#pragma unroll
    for (int t = 0; t < 4; ++t) {
      const float* wrow = W + (size_t)(n0 + t * 16 + row) * K;
      v16h b;
      load8_f32_cvt(wrow + k0 + off,      b, 0);
      load8_f32_cvt(wrow + k0 + 16 + off, b, 8);
      acc[t] = __builtin_amdgcn_wmma_f32_16x16x32_f16(
          false, a, false, b, (short)0, acc[t], false, false);
    }
  }
#pragma unroll
  for (int t = 0; t < 4; ++t) {
    const float bn = bias[n0 + t * 16 + row];
#pragma unroll
    for (int r = 0; r < 8; ++r) {
      const int m = m0 + r + hi * 8;
      Y[(size_t)m * N + n0 + t * 16 + row] = acc[t][r] + bn;
    }
  }
}

// ---------------------------------------------------------------------------
extern "C" void kernel_launch(void* const* d_in, const int* in_sizes, int n_in,
                              void* d_out, int out_size, void* d_ws, size_t ws_size,
                              hipStream_t stream) {
  const float* q    = (const float*)d_in[0];
  const float* k    = (const float*)d_in[1];
  const float* v    = (const float*)d_in[2];
  const float* mask = (const float*)d_in[3];
  const float* wq_w = (const float*)d_in[4];
  const float* wq_b = (const float*)d_in[5];
  const float* wk_w = (const float*)d_in[6];
  const float* wk_b = (const float*)d_in[7];
  const float* wv_w = (const float*)d_in[8];
  const float* wv_b = (const float*)d_in[9];
  const float* pl_w = (const float*)d_in[10];
  const float* pl_b = (const float*)d_in[11];
  float* out = (float*)d_out;

  // workspace: Qp | Kp | Vp | attn  (each 4096*1024 f16 = 8MB; 32MB total)
  const size_t elems = (size_t)MTOT * DMODEL;
  half_t* Qp = (half_t*)d_ws;
  half_t* Kp = Qp + elems;
  half_t* Vp = Kp + elems;
  half_t* Ap = Vp + elems;

  dim3 ggemm(DMODEL / 64, MTOT / 16);  // (16, 256) one wave32 per 16x64 strip
  proj_gemm_f32<<<ggemm, 32, 0, stream>>>(q, wq_w, wq_b, Qp, MTOT, DMODEL, DMODEL);
  proj_gemm_f32<<<ggemm, 32, 0, stream>>>(k, wk_w, wk_b, Kp, MTOT, DMODEL, DMODEL);
  proj_gemm_f32<<<ggemm, 32, 0, stream>>>(v, wv_w, wv_b, Vp, MTOT, DMODEL, DMODEL);

  attn_kernel<<<BATCH * NHEADS * (SEQ / 16), 32, 0, stream>>>(Qp, Kp, Vp, mask, Ap);

  out_gemm_f16<<<ggemm, 32, 0, stream>>>(Ap, pl_w, pl_b, out, MTOT, DMODEL, DMODEL);
}
